// us_conv_68332929679861
// MI455X (gfx1250) — matrix operand order repaired and checked by reading, not compiled, not run on hardware
//
#include <hip/hip_runtime.h>
#include <hip/hip_bf16.h>
#include <math.h>

typedef __attribute__((ext_vector_type(16))) _Float16 v16h;
typedef __attribute__((ext_vector_type(8)))  _Float16 v8h;
typedef __attribute__((ext_vector_type(8)))  float    v8f;

#define HH 512
#define WW 512
#define CIN 64
#define COUT 64
#define PADW 514   // (512 + 2) zero-padded for SAME 3x3

// ---------------------------------------------------------------------------
// Kernel 1: convert x (fp32 HWC) to f16 with 1-pixel zero border.
// ---------------------------------------------------------------------------
__global__ __launch_bounds__(256) void pad_convert(const float* __restrict__ x,
                                                   _Float16* __restrict__ xh) {
  int idx = blockIdx.x * blockDim.x + threadIdx.x;   // 0 .. 514*514-1
  if (idx >= PADW * PADW) return;
  int r = idx / PADW, c = idx % PADW;
  _Float16* dst = xh + (size_t)idx * CIN;
  if (r == 0 || r == PADW - 1 || c == 0 || c == PADW - 1) {
    #pragma unroll 8
    for (int k = 0; k < CIN; ++k) dst[k] = (_Float16)0.0f;
  } else {
    const float* src = x + ((size_t)(r - 1) * WW + (c - 1)) * CIN;
    #pragma unroll 8
    for (int k = 0; k < CIN; ++k) dst[k] = (_Float16)src[k];
  }
}

// ---------------------------------------------------------------------------
// Kernel 2: fold 1x1 conv into center tap, pre-swizzle weights into the
// CDNA5 16-bit B-matrix (32x16) register layout:
//   lane L, element e  <->  K = (L>>4)*8 + (e&7) + (e>>3)*16,  N = L&15
// Packed as [tap][kc][n0][lane][e]  (9*2*4*32*16 halfs).
// Also biasEff = b1 + b2.
// ---------------------------------------------------------------------------
__global__ __launch_bounds__(256) void pack_w(const float* __restrict__ w1,
                                              const float* __restrict__ w2,
                                              const float* __restrict__ b1,
                                              const float* __restrict__ b2,
                                              _Float16* __restrict__ wp,
                                              float* __restrict__ biasEff) {
  int idx = blockIdx.x * blockDim.x + threadIdx.x;   // 0 .. 36863
  if (idx < COUT) biasEff[idx] = b1[idx] + b2[idx];
  if (idx >= 9 * 2 * 4 * 32 * 16) return;
  int grp  = idx >> 9;          // (t*2+kc)*4 + n0 : 0..71
  int win  = idx & 511;
  int lane = win >> 4;
  int e    = win & 15;
  int t    = grp >> 3;
  int kc   = (grp >> 2) & 1;
  int n0   = grp & 3;
  int k    = kc * 32 + (lane >> 4) * 8 + (e & 7) + (e >> 3) * 16;
  int co   = n0 * 16 + (lane & 15);
  float v  = w1[((size_t)t * CIN + k) * COUT + co];
  if (t == 4) v += w2[(size_t)k * COUT + co];       // fold 1x1 conv into center tap
  wp[idx] = (_Float16)v;
}

// ---------------------------------------------------------------------------
// Kernel 3: implicit-GEMM 3x3 conv with v_wmma_f32_16x16x32_f16.
// Block = 256 threads = 8 waves; wave (pt, n0) computes a 16-pixel x 16-cout
// tile; 9 taps x 2 K-chunks = 18 WMMA per wave, f32 accumulate.
// ---------------------------------------------------------------------------
__global__ __launch_bounds__(256) void conv_wmma(const _Float16* __restrict__ xh,
                                                 const _Float16* __restrict__ wp,
                                                 const float* __restrict__ biasEff,
                                                 float* __restrict__ y) {
  const int tid   = threadIdx.x;
  const int wave  = tid >> 5;
  const int lane  = tid & 31;
  const int pt    = wave >> 2;       // which 16-pixel tile (0/1)
  const int n0    = wave & 3;        // which 16-cout tile (0..3)
  const int bid   = blockIdx.x;      // 0 .. 8191
  const int hrow  = bid >> 4;        // 0 .. 511
  const int wseg  = bid & 15;        // 0 .. 15
  const int w0    = wseg * 32 + pt * 16;
  const int m     = lane & 15;       // A-matrix row (pixel within tile)
  const int khalf = lane >> 4;       // which K half this lane holds

  v8f acc = {};
  const _Float16* wbase = wp + (size_t)n0 * 512 + (size_t)lane * 16;

  #pragma unroll
  for (int t = 0; t < 9; ++t) {
    const int dh = t / 3 - 1, dw = t % 3 - 1;
    const int prow = hrow + 1 + dh;
    const int pcol = w0 + 1 + dw + m;
    const _Float16* arow = xh + ((size_t)prow * PADW + pcol) * CIN;
    #pragma unroll
    for (int kc = 0; kc < 2; ++kc) {
      // 16-bit A layout: lane holds K = [kb .. kb+7] and [kb+16 .. kb+23]
      const int kb = kc * 32 + khalf * 8;
      v8h lo = *(const v8h*)(arow + kb);
      v8h hi = *(const v8h*)(arow + kb + 16);
      v16h a;
      #pragma unroll
      for (int e = 0; e < 8; ++e) { a[e] = lo[e]; a[e + 8] = hi[e]; }
      v16h b = *(const v16h*)(wbase + (size_t)(t * 2 + kc) * 2048);
      acc = __builtin_amdgcn_wmma_f32_16x16x32_f16(
          /*neg_a=*/false, a, /*neg_b=*/false, b,
          /*c_mod=*/(short)0, acc, /*reuse_a=*/false, /*reuse_b=*/false);
    }
  }

  // C/D layout: element r -> pixel M = r + 8*khalf, column N = lane&15
  const int n  = n0 * 16 + (lane & 15);
  const float bv = biasEff[n];
  float* ybase = y + ((size_t)(hrow * WW + w0 + 8 * khalf) * COUT) + n;
  #pragma unroll
  for (int r = 0; r < 8; ++r) ybase[(size_t)r * COUT] = acc[r] + bv;
}

// ---------------------------------------------------------------------------
// Kernel 4: transposed 2x bilinear upsample + per-row GroupNorm + tanh-GELU.
// One block per output row i.  up[i][j][c] = y(row a=j*512/1023, col b=i*512/1023),
// zero outside [0,512).  Stats per (i, group) over 1024 cols x 2 channels.
// Row staged in 256 KB dynamic LDS; stats via ds_add_f32 atomics.
// ---------------------------------------------------------------------------
__global__ __launch_bounds__(256) void upsample_gn_gelu(const float* __restrict__ y,
                                                        const float* __restrict__ gamma,
                                                        const float* __restrict__ beta,
                                                        float* __restrict__ out) {
  extern __shared__ float row[];          // 1024 * 64 floats = 256 KB
  __shared__ float gsum[32], gsum2[32];
  const int i   = blockIdx.x;             // output row (GroupNorm batch axis)
  const int tid = threadIdx.x;
  const int c   = tid & 63;               // fixed channel per thread
  const int g   = c >> 1;                 // group (2 channels/group)
  if (tid < 32) { gsum[tid] = 0.0f; gsum2[tid] = 0.0f; }
  __syncthreads();

  const float scale = 512.0f / 1023.0f;
  const float b  = (float)i * scale;
  const int   c0 = (int)floorf(b);
  const float fb = b - (float)c0;
  const bool  vc0 = (c0 < WW);
  const bool  vc1 = (c0 + 1 < WW);
  const int   cc0 = vc0 ? c0 : 0;
  const int   cc1 = vc1 ? c0 + 1 : 0;

  float s = 0.0f, s2 = 0.0f;
  for (int j = tid >> 6; j < 1024; j += 4) {
    const float a  = (float)j * scale;
    const int   r0 = (int)floorf(a);
    const float fa = a - (float)r0;
    const bool  vr0 = (r0 < HH);
    const bool  vr1 = (r0 + 1 < HH);
    const int   rr0 = vr0 ? r0 : 0;
    const int   rr1 = vr1 ? r0 + 1 : 0;
    const float y00 = (vr0 && vc0) ? y[((size_t)rr0 * WW + cc0) * COUT + c] : 0.0f;
    const float y01 = (vr0 && vc1) ? y[((size_t)rr0 * WW + cc1) * COUT + c] : 0.0f;
    const float y10 = (vr1 && vc0) ? y[((size_t)rr1 * WW + cc0) * COUT + c] : 0.0f;
    const float y11 = (vr1 && vc1) ? y[((size_t)rr1 * WW + cc1) * COUT + c] : 0.0f;
    const float v = (1.0f - fa) * ((1.0f - fb) * y00 + fb * y01) +
                    fa          * ((1.0f - fb) * y10 + fb * y11);
    row[j * 64 + c] = v;
    s += v; s2 += v * v;
  }
  atomicAdd(&gsum[g], s);                 // ds_add_f32
  atomicAdd(&gsum2[g], s2);
  __syncthreads();

  const float inv  = 1.0f / 2048.0f;      // 1024 cols * 2 channels
  const float mu   = gsum[g] * inv;
  const float var  = fmaxf(gsum2[g] * inv - mu * mu, 0.0f);
  const float rstd = rsqrtf(var + 1e-6f);
  const float gm = gamma[c], bt = beta[c];
  float* obase = out + (size_t)i * 1024 * 64;
  for (int j = tid >> 6; j < 1024; j += 4) {
    const float v = (row[j * 64 + c] - mu) * rstd * gm + bt;
    const float u = 0.7978845608028654f * (v + 0.044715f * v * v * v);
    obase[j * 64 + c] = 0.5f * v * (1.0f + tanhf(u));
  }
}

// ---------------------------------------------------------------------------
extern "C" void kernel_launch(void* const* d_in, const int* in_sizes, int n_in,
                              void* d_out, int out_size, void* d_ws, size_t ws_size,
                              hipStream_t stream) {
  const float* x     = (const float*)d_in[0];
  const float* w1    = (const float*)d_in[1];
  const float* b1    = (const float*)d_in[2];
  const float* w2    = (const float*)d_in[3];
  const float* b2    = (const float*)d_in[4];
  const float* gamma = (const float*)d_in[5];
  const float* beta  = (const float*)d_in[6];
  float* out = (float*)d_out;

  char* ws = (char*)d_ws;
  _Float16* xh = (_Float16*)ws;                       // 514*514*64*2 = 33,817,088 B
  size_t off = (size_t)PADW * PADW * CIN * 2;
  _Float16* wp = (_Float16*)(ws + off); off += 9 * 2 * 4 * 32 * 16 * 2;  // 73,728 B
  float* biasEff = (float*)(ws + off);  off += 256;
  float* y = (float*)(ws + off);                      // 512*512*64*4 = 67,108,864 B

  pad_convert<<<(PADW * PADW + 255) / 256, 256, 0, stream>>>(x, xh);
  pack_w<<<(9 * 2 * 4 * 32 * 16 + 255) / 256, 256, 0, stream>>>(w1, w2, b1, b2, wp, biasEff);
  conv_wmma<<<8192, 256, 0, stream>>>(xh, wp, biasEff, y);
  upsample_gn_gelu<<<1024, 256, 1024 * 64 * 4, stream>>>(y, gamma, beta, out);
}
